// Decoder_27736898798106
// MI455X (gfx1250) — matrix operand order repaired and checked by reading, not compiled, not run on hardware
//
#include <hip/hip_runtime.h>
#include <hip/hip_bf16.h>

// ---------------- problem constants ----------------
#define Bn     32
#define SRCn   400
#define TGTn   32
#define STEPS  31          // TGT-1
#define EMBn   256
#define HIDn   512
#define ATTn   512
#define VOCABn 50000
#define OOVn   500
#define VEXTn  50500
#define NEGC   (-1000000.0f)

typedef __bf16 bf16_t;
typedef __attribute__((ext_vector_type(8)))  __bf16 v8bf;
typedef __attribute__((ext_vector_type(16))) __bf16 v16bf;
typedef __attribute__((ext_vector_type(8)))  float  v8f;

#if defined(__HIP_DEVICE_COMPILE__)
typedef __attribute__((address_space(1))) int gas_int;   // global (prints as __device__ int*)
typedef __attribute__((address_space(3))) int las_int;   // LDS    (prints as __shared__ int*)
#endif

__device__ __forceinline__ bf16_t f2bf(float x) {
    union { float f; unsigned u; } v; v.f = x;
    unsigned r = v.u + 0x7FFFu + ((v.u >> 16) & 1u);   // round-to-nearest-even
    unsigned short hb = (unsigned short)(r >> 16);
    bf16_t out;
    __builtin_memcpy(&out, &hb, 2);
    return out;
}

__device__ __forceinline__ void atomicMaxF(float* addr, float val) {
    unsigned int* ua = (unsigned int*)addr;
    unsigned int old = *ua;
    while (true) {
        float f = __uint_as_float(old);
        if (f >= val) break;
        unsigned int assumed = old;
        old = atomicCAS(ua, assumed, __float_as_uint(val));
        if (old == assumed) break;
    }
}

// ---------------- bf16 WMMA GEMM:  C[M,N] = A[M,K] * W[N,K]^T + bias ----------------
// One wave computes a 32-row x 16-col strip: two 16x16 accumulators share one B
// fragment per k-step (halves B traffic vs tile-per-wave).
// A: bf16 row-major (lda), W: bf16 row-major N x K (ldb). C: f32 (ldc stride),
// optional bf16 mirror. Requires M % 32 == 0, N % 16 == 0, K % 32 == 0.
__global__ void k_gemm_bf16(const bf16_t* __restrict__ A, int lda,
                            const bf16_t* __restrict__ W, int ldb,
                            const float* __restrict__ bias,
                            float* __restrict__ C, int ldc,
                            bf16_t* __restrict__ Cbf,
                            int M, int N, int K)
{
    const int wavesPerBlock = blockDim.x >> 5;
    const int wid  = blockIdx.x * wavesPerBlock + (threadIdx.x >> 5);
    const int lane = threadIdx.x & 31;
    const int tilesN = N >> 4;
    const int strips = (M >> 5) * tilesN;
    if (wid >= strips) return;                     // uniform per-wave: EXEC stays full
    const int m0 = (wid / tilesN) << 5;            // 32-row strip
    const int n0 = (wid % tilesN) << 4;
    const int hl = lane >> 4;                      // lane half (K-split)
    const int l  = lane & 15;

    v8f c0 = {0.f,0.f,0.f,0.f,0.f,0.f,0.f,0.f};
    v8f c1 = {0.f,0.f,0.f,0.f,0.f,0.f,0.f,0.f};
    const bf16_t* arow0 = A + (size_t)(m0 + l)      * (size_t)lda;
    const bf16_t* arow1 = A + (size_t)(m0 + 16 + l) * (size_t)lda;
    const bf16_t* brow  = W + (size_t)(n0 + l)      * (size_t)ldb;

    for (int k = 0; k < K; k += 32) {
        // A 16x32 frag: lane l = row; hl=0 -> K[k..k+7]&K[k+16..k+23], hl=1 -> +8
        v8bf a0lo = *(const v8bf*)(arow0 + k + hl * 8);
        v8bf a0hi = *(const v8bf*)(arow0 + k + 16 + hl * 8);
        v8bf a1lo = *(const v8bf*)(arow1 + k + hl * 8);
        v8bf a1hi = *(const v8bf*)(arow1 + k + 16 + hl * 8);
        // B 32x16 frag: lane l = column; hl=0 -> K[k..k+7]&K[k+16..k+23], hl=1 -> +8
        v8bf blo = *(const v8bf*)(brow + k + hl * 8);
        v8bf bhi = *(const v8bf*)(brow + k + 16 + hl * 8);
        v16bf a0, a1, b;
#pragma unroll
        for (int i = 0; i < 8; ++i) {
            a0[i] = a0lo[i]; a0[i + 8] = a0hi[i];
            a1[i] = a1lo[i]; a1[i + 8] = a1hi[i];
            b[i]  = blo[i];  b[i + 8]  = bhi[i];
        }
        c0 = __builtin_amdgcn_wmma_f32_16x16x32_bf16(false, a0, false, b,
                                                     (short)0, c0, false, false);
        c1 = __builtin_amdgcn_wmma_f32_16x16x32_bf16(false, a1, false, b,
                                                     (short)0, c1, false, false);
    }

    const int col = n0 + l;
    const float bv = bias ? bias[col] : 0.0f;
#pragma unroll
    for (int i = 0; i < 8; ++i) {
        const int r0 = m0 + hl * 8 + i;            // C layout: vgpr i -> M = i (+8 hi half)
        const int r1 = r0 + 16;
        const float v0 = c0[i] + bv;
        const float v1 = c1[i] + bv;
        C[(size_t)r0 * (size_t)ldc + col] = v0;
        C[(size_t)r1 * (size_t)ldc + col] = v1;
        if (Cbf) {
            Cbf[(size_t)r0 * (size_t)ldc + col] = f2bf(v0);
            Cbf[(size_t)r1 * (size_t)ldc + col] = f2bf(v1);
        }
    }
}

// ---------------- elementwise / helper kernels ----------------
__global__ void k_f2bf_arr(const float* __restrict__ s, bf16_t* __restrict__ d, int n) {
    for (int i = blockIdx.x * blockDim.x + threadIdx.x; i < n; i += gridDim.x * blockDim.x)
        d[i] = f2bf(s[i]);
}

__global__ void k_init_state(const float* __restrict__ h0, float* __restrict__ h,
                             bf16_t* __restrict__ hbf, float* __restrict__ ctxt) {
    int i = blockIdx.x * blockDim.x + threadIdx.x;
    if (i >= Bn * HIDn) return;
    float v = h0[i];
    h[i] = v; hbf[i] = f2bf(v); ctxt[i] = 0.0f;
}

__global__ void k_concat2(const float* __restrict__ tgt_t, const float* __restrict__ ctxt,
                          bf16_t* __restrict__ A1) {
    int i = blockIdx.x * blockDim.x + threadIdx.x;
    if (i >= Bn * (EMBn + HIDn)) return;
    int b = i / (EMBn + HIDn), k = i % (EMBn + HIDn);
    float v = (k < EMBn) ? tgt_t[b * EMBn + k] : ctxt[b * HIDn + (k - EMBn)];
    A1[i] = f2bf(v);
}

__global__ void k_concat3(const float* __restrict__ tgt_t, const float* __restrict__ ctxt,
                          const float* __restrict__ h, bf16_t* __restrict__ A3) {
    const int W = EMBn + 2 * HIDn;                 // 1280
    int i = blockIdx.x * blockDim.x + threadIdx.x;
    if (i >= Bn * W) return;
    int b = i / W, k = i % W;
    float v;
    if (k < EMBn)              v = tgt_t[b * EMBn + k];
    else if (k < EMBn + HIDn)  v = ctxt[b * HIDn + (k - EMBn)];
    else                       v = h[b * HIDn + (k - EMBn - HIDn)];
    A3[i] = f2bf(v);
}

__global__ void k_gru(const float* __restrict__ gi, const float* __restrict__ gh,
                      float* __restrict__ h, bf16_t* __restrict__ hbf) {
    int i = blockIdx.x * blockDim.x + threadIdx.x;
    if (i >= Bn * HIDn) return;
    int b = i / HIDn, j = i % HIDn;
    const float* gib = gi + (size_t)b * 3 * HIDn;
    const float* ghb = gh + (size_t)b * 3 * HIDn;
    float r = 1.0f / (1.0f + expf(-(gib[j] + ghb[j])));
    float z = 1.0f / (1.0f + expf(-(gib[HIDn + j] + ghb[HIDn + j])));
    float n = tanhf(gib[2 * HIDn + j] + r * ghb[2 * HIDn + j]);
    float hv = (1.0f - z) * n + z * h[i];
    h[i] = hv; hbf[i] = f2bf(hv);
}

// e[b,s] = sum_a tanh(memAttn[b,s,a] + q[b,a]) * w_e[a], then mask.
// 8 waves/block, one (b,s) per wave; q row staged to LDS via async-to-LDS path.
__global__ void k_energy(const float* __restrict__ memAttn, const float* __restrict__ q,
                         const float* __restrict__ we, const float* __restrict__ mmask,
                         float* __restrict__ e_ws, float* __restrict__ e_out) {
    __shared__ float qs[ATTn];
    const int b  = blockIdx.x / (SRCn / 8);
    const int s0 = (blockIdx.x % (SRCn / 8)) * 8;
    const int tid = threadIdx.x;

#if defined(__HIP_DEVICE_COMPILE__) && __has_builtin(__builtin_amdgcn_global_load_async_to_lds_b32)
    for (int i = tid; i < ATTn; i += 256) {
        const float* gp = q + b * ATTn + i;
        __builtin_amdgcn_global_load_async_to_lds_b32(
            (gas_int*)(unsigned long long)(const void*)gp,
            (las_int*)(unsigned)(unsigned long long)(const void*)&qs[i],
            0, 0);
    }
#if __has_builtin(__builtin_amdgcn_s_wait_asynccnt)
    __builtin_amdgcn_s_wait_asynccnt(0);
#else
    asm volatile("s_wait_asynccnt 0" ::: "memory");
#endif
#else
    for (int i = tid; i < ATTn; i += 256) qs[i] = q[b * ATTn + i];
#endif
    __syncthreads();

    const int w = tid >> 5, lane = tid & 31;
    const int s = s0 + w;
    const float* row = memAttn + ((size_t)(b * SRCn + s)) * ATTn;
    float acc = 0.0f;
    for (int i = lane; i < ATTn; i += 32) acc += tanhf(row[i] + qs[i]) * we[i];
    for (int o = 16; o > 0; o >>= 1) acc += __shfl_down(acc, o, 32);
    if (lane == 0) {
        float mk = mmask[b * SRCn + s];
        float ev = acc * mk + (1.0f - mk) * NEGC;
        e_ws[b * SRCn + s]  = ev;
        e_out[b * SRCn + s] = ev;
    }
}

// block per batch row: softmax over SRC then ctxt[b,h] = sum_s score[s]*memory[b,s,h]
__global__ void k_softmax_ctxt(const float* __restrict__ e, const float* __restrict__ memory,
                               float* __restrict__ ctxt) {
    __shared__ float sc[SRCn];
    __shared__ float red[256];
    const int b = blockIdx.x, tid = threadIdx.x;
    for (int i = tid; i < SRCn; i += 256) sc[i] = e[b * SRCn + i];
    __syncthreads();
    float lm = -3.4e38f;
    for (int i = tid; i < SRCn; i += 256) lm = fmaxf(lm, sc[i]);
    red[tid] = lm; __syncthreads();
    for (int o = 128; o > 0; o >>= 1) { if (tid < o) red[tid] = fmaxf(red[tid], red[tid + o]); __syncthreads(); }
    const float m = red[0]; __syncthreads();
    float ls = 0.0f;
    for (int i = tid; i < SRCn; i += 256) { float v = expf(sc[i] - m); sc[i] = v; ls += v; }
    red[tid] = ls; __syncthreads();
    for (int o = 128; o > 0; o >>= 1) { if (tid < o) red[tid] += red[tid + o]; __syncthreads(); }
    const float inv = 1.0f / red[0];
    __syncthreads();
    for (int h = tid; h < HIDn; h += 256) {
        float acc = 0.0f;
        const float* mp = memory + (size_t)b * SRCn * HIDn + h;
        for (int s = 0; s < SRCn; ++s) acc += sc[s] * mp[(size_t)s * HIDn];
        ctxt[b * HIDn + h] = acc * inv;
    }
}

__global__ void k_maxout(const float* __restrict__ r, bf16_t* __restrict__ m) {
    int i = blockIdx.x * blockDim.x + threadIdx.x;
    if (i >= Bn * (HIDn / 2)) return;
    int b = i / (HIDn / 2), j = i % (HIDn / 2);
    m[i] = f2bf(fmaxf(r[b * HIDn + 2 * j], r[b * HIDn + 2 * j + 1]));
}

__global__ void k_fillneg(float* __restrict__ cp) {
    const int n = Bn * VEXTn;
    for (int i = blockIdx.x * blockDim.x + threadIdx.x; i < n; i += gridDim.x * blockDim.x)
        cp[i] = NEGC;
}

__global__ void k_scatter(const int* __restrict__ idx, const float* __restrict__ e,
                          float* __restrict__ cp) {
    int i = blockIdx.x * blockDim.x + threadIdx.x;
    if (i >= Bn * SRCn) return;
    int b = i / SRCn;
    atomicMaxF(cp + (size_t)b * VEXTn + idx[i], e[i]);
}

// restore NEG only at the scattered positions (instead of refilling 6.4 MB)
__global__ void k_reset_copy(const int* __restrict__ idx, float* __restrict__ cp) {
    int i = blockIdx.x * blockDim.x + threadIdx.x;
    if (i >= Bn * SRCn) return;
    int b = i / SRCn;
    cp[(size_t)b * VEXTn + idx[i]] = NEGC;
}

// final = gen_ext + copy, with copy==NEG -> 0; OOV pad implied zero for gen.
__global__ void k_finalize(float* __restrict__ out_t, const float* __restrict__ cp) {
    const int n = Bn * VEXTn;
    for (int i = blockIdx.x * blockDim.x + threadIdx.x; i < n; i += gridDim.x * blockDim.x) {
        int v = i % VEXTn;
        float c = cp[i];
        c = (c == NEGC) ? 0.0f : c;
        float g = (v < VOCABn) ? out_t[i] : 0.0f;
        float f = g + c;
        if (f == 0.0f && v >= VOCABn) f = NEGC;
        out_t[i] = f;
    }
}

// ---------------- host side ----------------
static inline void launch_gemm(const bf16_t* A, int lda, const bf16_t* W, int ldb,
                               const float* bias, float* C, int ldc, bf16_t* Cbf,
                               int M, int N, int K, hipStream_t s) {
    int waves = (M / 32) * (N / 16);
    int blocks = (waves + 3) / 4;
    k_gemm_bf16<<<blocks, 128, 0, s>>>(A, lda, W, ldb, bias, C, ldc, Cbf, M, N, K);
}

extern "C" void kernel_launch(void* const* d_in, const int* in_sizes, int n_in,
                              void* d_out, int out_size, void* d_ws, size_t ws_size,
                              hipStream_t stream) {
    (void)in_sizes; (void)n_in; (void)out_size; (void)ws_size;

    const float* memory = (const float*)d_in[0];
    const float* mmask  = (const float*)d_in[1];
    const int*   srcIdx = (const int*)d_in[2];
    const float* tgt    = (const float*)d_in[3];
    const float* h0     = (const float*)d_in[4];
    const float* W_mem  = (const float*)d_in[5];
    const float* b_mem  = (const float*)d_in[6];
    const float* W_q    = (const float*)d_in[7];
    const float* w_e    = (const float*)d_in[8];
    const float* W_gi   = (const float*)d_in[9];
    const float* b_gi   = (const float*)d_in[10];
    const float* W_ih   = (const float*)d_in[11];
    const float* b_ih   = (const float*)d_in[12];
    const float* W_hh   = (const float*)d_in[13];
    const float* b_hh   = (const float*)d_in[14];
    const float* W_r    = (const float*)d_in[15];
    const float* b_r    = (const float*)d_in[16];
    const float* W_gen  = (const float*)d_in[17];
    const float* b_gen  = (const float*)d_in[18];

    float* outGen = (float*)d_out;                               // (31, 32, 50500)
    float* outE   = outGen + (size_t)STEPS * Bn * VEXTn;         // (31, 32, 400)

    // ---- workspace carve (256B aligned) ----
    size_t off = 0;
    auto carve = [&](size_t bytes) -> void* {
        void* r = (char*)d_ws + off;
        off += (bytes + 255) & ~(size_t)255;
        return r;
    };
    float*  memAttn = (float*) carve((size_t)Bn * SRCn * ATTn * 4);
    bf16_t* memBf   = (bf16_t*)carve((size_t)Bn * SRCn * HIDn * 2);
    bf16_t* WgenBf  = (bf16_t*)carve((size_t)VOCABn * EMBn * 2);     // 50000 x 256
    bf16_t* WmemBf  = (bf16_t*)carve((size_t)ATTn * HIDn * 2);
    bf16_t* WgiBf   = (bf16_t*)carve((size_t)HIDn * (EMBn + HIDn) * 2);
    bf16_t* WihBf   = (bf16_t*)carve((size_t)3 * HIDn * HIDn * 2);
    bf16_t* WhhBf   = (bf16_t*)carve((size_t)3 * HIDn * HIDn * 2);
    bf16_t* WqBf    = (bf16_t*)carve((size_t)ATTn * HIDn * 2);
    bf16_t* WrBf    = (bf16_t*)carve((size_t)HIDn * (EMBn + 2 * HIDn) * 2);
    float*  copyBuf = (float*) carve((size_t)Bn * VEXTn * 4);
    float*  hBuf    = (float*) carve((size_t)Bn * HIDn * 4);
    bf16_t* hBf     = (bf16_t*)carve((size_t)Bn * HIDn * 2);
    float*  ctxtBuf = (float*) carve((size_t)Bn * HIDn * 4);
    float*  gruIn   = (float*) carve((size_t)Bn * HIDn * 4);
    bf16_t* gruInBf = (bf16_t*)carve((size_t)Bn * HIDn * 2);
    float*  giBuf   = (float*) carve((size_t)Bn * 3 * HIDn * 4);
    float*  ghBuf   = (float*) carve((size_t)Bn * 3 * HIDn * 4);
    float*  qBuf    = (float*) carve((size_t)Bn * ATTn * 4);
    float*  eBuf    = (float*) carve((size_t)Bn * SRCn * 4);
    bf16_t* A1Bf    = (bf16_t*)carve((size_t)Bn * (EMBn + HIDn) * 2);
    bf16_t* A3Bf    = (bf16_t*)carve((size_t)Bn * (EMBn + 2 * HIDn) * 2);
    float*  rtBuf   = (float*) carve((size_t)Bn * HIDn * 4);
    bf16_t* mBf     = (bf16_t*)carve((size_t)Bn * (HIDn / 2) * 2);

    auto cvt = [&](const float* s, bf16_t* d, int n) {
        int blocks = (n + 255) / 256; if (blocks > 4096) blocks = 4096;
        k_f2bf_arr<<<blocks, 256, 0, stream>>>(s, d, n);
    };

    // ---- one-time precompute ----
    cvt(memory, memBf, Bn * SRCn * HIDn);
    cvt(W_gen,  WgenBf, VOCABn * EMBn);
    cvt(W_mem,  WmemBf, ATTn * HIDn);
    cvt(W_gi,   WgiBf,  HIDn * (EMBn + HIDn));
    cvt(W_ih,   WihBf,  3 * HIDn * HIDn);
    cvt(W_hh,   WhhBf,  3 * HIDn * HIDn);
    cvt(W_q,    WqBf,   ATTn * HIDn);
    cvt(W_r,    WrBf,   HIDn * (EMBn + 2 * HIDn));
    k_init_state<<<(Bn * HIDn + 255) / 256, 256, 0, stream>>>(h0, hBuf, hBf, ctxtBuf);
    k_fillneg<<<4096, 256, 0, stream>>>(copyBuf);                // once; reset incrementally

    // mem_attn[b,s,:] = memory[b,s,:] @ W_mem^T + b_mem   (M=12800, N=512, K=512)
    launch_gemm(memBf, HIDn, WmemBf, HIDn, b_mem, memAttn, ATTn, nullptr,
                Bn * SRCn, ATTn, HIDn, stream);

    // ---- sequential decode ----
    for (int t = 0; t < STEPS; ++t) {
        const float* tgt_t = tgt + (size_t)t * Bn * EMBn;
        float* out_t = outGen + (size_t)t * Bn * VEXTn;
        float* e_out = outE + (size_t)t * Bn * SRCn;

        // gru_in = [tgt, ctxt] @ W_gi^T + b_gi
        k_concat2<<<(Bn * (EMBn + HIDn) + 255) / 256, 256, 0, stream>>>(tgt_t, ctxtBuf, A1Bf);
        launch_gemm(A1Bf, EMBn + HIDn, WgiBf, EMBn + HIDn, b_gi, gruIn, HIDn, gruInBf,
                    Bn, HIDn, EMBn + HIDn, stream);

        // GRU gates
        launch_gemm(gruInBf, HIDn, WihBf, HIDn, b_ih, giBuf, 3 * HIDn, nullptr,
                    Bn, 3 * HIDn, HIDn, stream);
        launch_gemm(hBf, HIDn, WhhBf, HIDn, b_hh, ghBuf, 3 * HIDn, nullptr,
                    Bn, 3 * HIDn, HIDn, stream);
        k_gru<<<(Bn * HIDn + 255) / 256, 256, 0, stream>>>(giBuf, ghBuf, hBuf, hBf);

        // attention
        launch_gemm(hBf, HIDn, WqBf, HIDn, nullptr, qBuf, ATTn, nullptr,
                    Bn, ATTn, HIDn, stream);
        k_energy<<<Bn * (SRCn / 8), 256, 0, stream>>>(memAttn, qBuf, w_e, mmask, eBuf, e_out);
        k_softmax_ctxt<<<Bn, 256, 0, stream>>>(eBuf, memory, ctxtBuf);

        // readout + maxout
        k_concat3<<<(Bn * (EMBn + 2 * HIDn) + 255) / 256, 256, 0, stream>>>(tgt_t, ctxtBuf, hBuf, A3Bf);
        launch_gemm(A3Bf, EMBn + 2 * HIDn, WrBf, EMBn + 2 * HIDn, b_r, rtBuf, HIDn, nullptr,
                    Bn, HIDn, EMBn + 2 * HIDn, stream);
        k_maxout<<<(Bn * (HIDn / 2) + 255) / 256, 256, 0, stream>>>(rtBuf, mBf);

        // gen projection straight into output slice (ldc = V_EXT)
        launch_gemm(mBf, EMBn, WgenBf, EMBn, b_gen, out_t, VEXTn, nullptr,
                    Bn, VOCABn, EMBn, stream);

        // copy mechanism
        k_scatter<<<(Bn * SRCn + 255) / 256, 256, 0, stream>>>(srcIdx, eBuf, copyBuf);
        k_finalize<<<4096, 256, 0, stream>>>(out_t, copyBuf);
        k_reset_copy<<<(Bn * SRCn + 255) / 256, 256, 0, stream>>>(srcIdx, copyBuf);
    }
}
